// Atlas_154618823086
// MI455X (gfx1250) — matrix operand order repaired and checked by reading, not compiled, not run on hardware
//
// MI455X (gfx1250) implementation of the Atlas chunked-TTT attention layer.
//
// Strategy (reasoned from MI455X specs, compile-only):
//  * The 5 big (16384x1024)x(1024x1024) projections dominate FLOPs (~172 GF).
//    Executed with v_wmma_f32_16x16x32_bf16 (wave32 WMMA), block tile
//    128x128x32, 8 waves/WG each owning 32x64 (8 WMMAs/wave/K-step),
//    LDS double-buffered with stride-40 padding.
//  * Global->LDS staging uses gfx1250 async DMA (global_load_async_to_lds_b128,
//    ASYNCcnt) synchronized with s_wait_asynccnt + workgroup barrier.
//  * The sequential chunk scan (256 steps, only b*H=64 parallel units, ~3 GF)
//    runs as a small VALU kernel - latency-bound, not compute-bound.
//  * Elementwise stages (conv+SiLU+L2, softplus LR, LN*gate) are
//    bandwidth-trivial at 23.3 TB/s.
// Workspace required: ~470 MB.

#include <hip/hip_runtime.h>
#include <hip/hip_bf16.h>
#include <stdint.h>

#define DIM     1024
#define NHEAD   16
#define HD      64
#define D_INNER 4
#define CHUNK   16
#define KSZ     4
#define BATCH   4
#define SEQ     4096
#define NROWS   (BATCH * SEQ)     // 16384
#define BASE_LR 0.001f

typedef __bf16 bf16;
typedef __attribute__((ext_vector_type(16))) __bf16 v16bf;
typedef __attribute__((ext_vector_type(8)))  __bf16 v8bf;
typedef __attribute__((ext_vector_type(8)))  float  v8f;

// ---------------------------------------------------------------- conversions
__global__ void cvt_bf16_kernel(const float* __restrict__ in,
                                bf16* __restrict__ out, int n) {
    int i = blockIdx.x * blockDim.x + threadIdx.x;
    int stride = gridDim.x * blockDim.x;
    for (; i < n; i += stride) out[i] = (bf16)in[i];
}

// ---------------------------------------------------------------- WMMA GEMM
// Y[M,N] = X[M,K] * W[N,K]^T   (i.e. out feature i uses row i of W)
#define BM 128
#define BN 128
#define BK 32
#define TS 40   // padded LDS stride (bf16 elems) to avoid bank conflicts

// gfx1250 async DMA: per-lane global -> LDS, tracked with ASYNCcnt.
__device__ __forceinline__ void async_g2l_b128(uint32_t lds_addr,
                                               const bf16* gaddr) {
    asm volatile("global_load_async_to_lds_b128 %0, %1, off"
                 :: "v"(lds_addr), "v"(gaddr) : "memory");
}
__device__ __forceinline__ void wait_asynccnt0() {
    asm volatile("s_wait_asynccnt 0" ::: "memory");
}

__global__ __launch_bounds__(256)
void gemm_bf16_wmma(const bf16* __restrict__ X, const bf16* __restrict__ W,
                    float* __restrict__ Y, int M, int Nn, int K) {
    __shared__ __align__(16) bf16 As[2][BM * TS];
    __shared__ __align__(16) bf16 Bs[2][BN * TS];

    const int tid  = threadIdx.x;
    const int wid  = tid >> 5;
    const int lane = tid & 31;
    const int half = lane >> 4;
    const int l16  = lane & 15;

    const int m0 = blockIdx.y * BM;
    const int n0 = blockIdx.x * BN;
    const int wm = (wid & 3) * 32;    // wave M offset inside block tile
    const int wn = (wid >> 2) * 64;   // wave N offset inside block tile

    // cooperative loader: each thread DMAs 2x16B of A and 2x16B of B
    const int lr = tid >> 1;          // 0..127 : row of A tile / row of B tile
    const int lk = (tid & 1) * 16;    // 0,16   : K offset

    v8f acc[2][4] = {};

    auto load_tile_async = [&](int buf, int kk) {
        const uint32_t ldsA =
            (uint32_t)(uintptr_t)&As[buf][lr * TS + lk];
        const bf16* ga = X + (size_t)(m0 + lr) * K + kk + lk;
        async_g2l_b128(ldsA,      ga);
        async_g2l_b128(ldsA + 16, ga + 8);
        const uint32_t ldsB =
            (uint32_t)(uintptr_t)&Bs[buf][lr * TS + lk];
        const bf16* gb = W + (size_t)(n0 + lr) * K + kk + lk;
        async_g2l_b128(ldsB,      gb);
        async_g2l_b128(ldsB + 16, gb + 8);
    };

    load_tile_async(0, 0);
    const int KT = K / BK;
    for (int kt = 0; kt < KT; ++kt) {
        const int cur = kt & 1;
        wait_asynccnt0();     // our DMA into buffer `cur` is complete
        __syncthreads();      // ... and everyone else's too
        if (kt + 1 < KT) {
            if (kt + 2 < KT)
                __builtin_prefetch(X + (size_t)(m0 + lr) * K + (kt + 2) * BK, 0, 1);
            load_tile_async(cur ^ 1, (kt + 1) * BK);
        }
        // A fragments: ISA 16-bit A 16x32 layout (K pairs split by lane half)
        v16bf afr[2], bfr[4];
        #pragma unroll
        for (int mi = 0; mi < 2; ++mi) {
            const bf16* ap = &As[cur][(wm + mi * 16 + l16) * TS];
            v8bf lo = *(const v8bf*)(ap + 8 * half);        // K = 8h .. 8h+7
            v8bf hi = *(const v8bf*)(ap + 16 + 8 * half);   // K = 16+8h ..
            #pragma unroll
            for (int e = 0; e < 8; ++e) { afr[mi][e] = lo[e]; afr[mi][e + 8] = hi[e]; }
        }
        // B fragments: col = lane&15, K = 8h..8h+7 then 16+8h.. (same as A)
        #pragma unroll
        for (int ni = 0; ni < 4; ++ni) {
            const bf16* bp = &Bs[cur][(wn + ni * 16 + l16) * TS];
            v8bf lo = *(const v8bf*)(bp + 8 * half);
            v8bf hi = *(const v8bf*)(bp + 16 + 8 * half);
            #pragma unroll
            for (int e = 0; e < 8; ++e) { bfr[ni][e] = lo[e]; bfr[ni][e + 8] = hi[e]; }
        }
        #pragma unroll
        for (int mi = 0; mi < 2; ++mi)
            #pragma unroll
            for (int ni = 0; ni < 4; ++ni)
                acc[mi][ni] = __builtin_amdgcn_wmma_f32_16x16x32_bf16(
                    false, afr[mi], false, bfr[ni], (short)0, acc[mi][ni],
                    false, false);
    }

    // C layout: VGPR r -> M = r + 8*half, N = lane&15
    #pragma unroll
    for (int mi = 0; mi < 2; ++mi)
        #pragma unroll
        for (int ni = 0; ni < 4; ++ni) {
            const int col = n0 + wn + ni * 16 + l16;
            const int row = m0 + wm + mi * 16 + half * 8;
            #pragma unroll
            for (int r = 0; r < 8; ++r)
                Y[(size_t)(row + r) * Nn + col] = acc[mi][ni][r];
        }
}

// ------------------------------------------- depthwise causal conv + SiLU(+L2)
__global__ __launch_bounds__(256)
void conv_silu_norm_kernel(const float* __restrict__ X,   // NROWS x DIM
                           const float* __restrict__ Wc,  // DIM x KSZ
                           float* __restrict__ Y, int do_l2norm) {
    __shared__ float hsum[NHEAD];
    const int n   = blockIdx.x;
    const int t   = n & (SEQ - 1);
    const int tid = threadIdx.x;
    if (tid < NHEAD) hsum[tid] = 0.f;
    __syncthreads();

    float val[4];
    #pragma unroll
    for (int i = 0; i < 4; ++i) {
        const int c = tid * 4 + i;
        float acc = X[(size_t)n * DIM + c];   // residual
        #pragma unroll
        for (int j = 0; j < KSZ; ++j) {
            const int tt = t - (KSZ - 1) + j;
            const float xv = (tt >= 0) ? X[(size_t)(n - (KSZ - 1) + j) * DIM + c] : 0.f;
            acc += Wc[c * KSZ + j] * xv;
        }
        val[i] = acc / (1.f + __expf(-acc));  // SiLU
    }
    if (do_l2norm) {
        const int h = (tid * 4) / HD;
        float ss = val[0]*val[0] + val[1]*val[1] + val[2]*val[2] + val[3]*val[3];
        atomicAdd(&hsum[h], ss);
        __syncthreads();
        const float inv = rsqrtf(hsum[h] + 1e-30f);
        #pragma unroll
        for (int i = 0; i < 4; ++i) val[i] *= inv;
    }
    #pragma unroll
    for (int i = 0; i < 4; ++i) Y[(size_t)n * DIM + tid * 4 + i] = val[i];
}

// ----------------------------------------------------------- LR head (N x 32)
__global__ __launch_bounds__(256)
void lr_kernel(const float* __restrict__ Hs, const float* __restrict__ Wlr,
               float* __restrict__ Lr) {
    __shared__ float red[256];
    const int n = blockIdx.x, tid = threadIdx.x;
    const int f = tid & 31, part = tid >> 5;
    const float* hrow = Hs + (size_t)n * DIM + part * 128;
    const float* wrow = Wlr + (size_t)f * DIM + part * 128;
    float s = 0.f;
    #pragma unroll 4
    for (int e = 0; e < 128; ++e) s += hrow[e] * wrow[e];
    red[tid] = s;
    __syncthreads();
    if (tid < 32) {
        float acc = 0.f;
        #pragma unroll
        for (int p = 0; p < 8; ++p) acc += red[p * 32 + tid];
        const float x = acc + BASE_LR;
        Lr[(size_t)n * 32 + tid] = (x > 20.f) ? x : log1pf(__expf(x));  // softplus
    }
}

// --------------------------------------------------- sequential chunked scan
__global__ __launch_bounds__(64)
void scan_kernel(const float* __restrict__ Q, const float* __restrict__ Kt,
                 const float* __restrict__ V, const float* __restrict__ Lr,
                 const float* __restrict__ Win0, const float* __restrict__ Wout0,
                 float* __restrict__ O) {
    const int bh = blockIdx.x;
    const int bb = bh / NHEAD, hh = bh % NHEAD;
    const int t  = threadIdx.x;  // 0..63 : column of the head dim

    __shared__ float sQ[CHUNK][HD], sK[CHUNK][HD], sV[CHUNK][HD];
    __shared__ float sWin[D_INNER][HD], sWout[D_INNER][HD];
    __shared__ float sKh[CHUNK][D_INNER], sQh[CHUNK][D_INNER];
    __shared__ float sQK[CHUNK][CHUNK];
    __shared__ float sS1[64], sS2[64];
    __shared__ float sLr[CHUNK][2];

    #pragma unroll
    for (int D = 0; D < D_INNER; ++D) {
        sWin[D][t]  = Win0[((size_t)D * NHEAD + hh) * HD + t];
        sWout[D][t] = Wout0[((size_t)D * NHEAD + hh) * HD + t];
    }
    const size_t base = (size_t)bb * SEQ * DIM + hh * HD;

    for (int c = 0; c < SEQ / CHUNK; ++c) {
        __syncthreads();
        #pragma unroll
        for (int l = 0; l < CHUNK; ++l) {
            const size_t off = base + (size_t)(c * CHUNK + l) * DIM + t;
            sQ[l][t] = Q[off]; sK[l][t] = Kt[off]; sV[l][t] = V[off];
        }
        if (t < CHUNK * 2) {
            const int l = t >> 1, j = t & 1;
            sLr[l][j] = Lr[((size_t)bb * SEQ + c * CHUNK + l) * 32 + hh * 2 + j];
        }
        __syncthreads();
        {   // k_t . W_in  and  q_t . W_in : 16x4 each, one dot per thread
            const int l = t >> 2, D = t & 3;
            float a = 0.f, b = 0.f;
            for (int d = 0; d < HD; ++d) { a += sK[l][d] * sWin[D][d];
                                           b += sQ[l][d] * sWin[D][d]; }
            sS1[t] = a; sS2[t] = b;
        }
        __syncthreads();
        if (t < CHUNK) {   // softmax over D=4
            float m1 = sS1[t*4], m2 = sS2[t*4];
            for (int D = 1; D < 4; ++D) { m1 = fmaxf(m1, sS1[t*4+D]);
                                          m2 = fmaxf(m2, sS2[t*4+D]); }
            float e1[4], e2[4], z1 = 0.f, z2 = 0.f;
            for (int D = 0; D < 4; ++D) {
                e1[D] = __expf(sS1[t*4+D] - m1); z1 += e1[D];
                e2[D] = __expf(sS2[t*4+D] - m2); z2 += e2[D];
            }
            const float lr1 = sLr[t][1];
            for (int D = 0; D < 4; ++D) { sKh[t][D] = e1[D] / z1 * lr1;
                                          sQh[t][D] = e2[D] / z2; }
        }
        __syncthreads();
        {   // masked qk (16x16)
            const int qr = t >> 2;
            for (int kc4 = 0; kc4 < 4; ++kc4) {
                const int kc = (t & 3) * 4 + kc4;
                float v = 0.f;
                for (int D = 0; D < 4; ++D) v += sQh[qr][D] * sKh[kc][D];
                sQK[qr][kc] = (kc <= qr) ? v : 0.f;
            }
        }
        __syncthreads();
        {   // o_t and rank-update of W_out (column-parallel, no cross hazards)
            float wout[4];
            for (int D = 0; D < 4; ++D) wout[D] = sWout[D][t];
            for (int l = 0; l < CHUNK; ++l) {
                float o = 0.f;
                for (int D = 0; D < 4; ++D) o += sQh[l][D] * wout[D];
                for (int k = 0; k <= l; ++k) o += sQK[l][k] * sV[k][t];
                O[base + (size_t)(c * CHUNK + l) * DIM + t] = o;
            }
            float upd[4] = {0.f, 0.f, 0.f, 0.f};
            for (int l = 0; l < CHUNK; ++l)
                for (int D = 0; D < 4; ++D) upd[D] += sKh[l][D] * sV[l][t];
            for (int D = 0; D < 4; ++D) sWout[D][t] = wout[D] + upd[D];
        }
        __syncthreads();
        const float lr_in = sLr[0][0], lr_out = sLr[0][1];
        for (int r = 0; r < 2; ++r) {
            {   // attn scores: W_in.k_t and W_out.v_t (4x16), /sqrt(64)
                const int D = t >> 4, l = t & 15;
                float a = 0.f, b = 0.f;
                for (int d = 0; d < HD; ++d) { a += sWin[D][d]  * sK[l][d];
                                               b += sWout[D][d] * sV[l][d]; }
                sS1[t] = a * 0.125f; sS2[t] = b * 0.125f;
            }
            __syncthreads();
            if (t < 8) {   // softmax over l=16 for both score sets
                const int D = t & 3;
                float* S = (t < 4) ? sS1 : sS2;
                float m = S[D*16];
                for (int l = 1; l < 16; ++l) m = fmaxf(m, S[D*16+l]);
                float z = 0.f;
                for (int l = 0; l < 16; ++l) { float e = __expf(S[D*16+l] - m);
                                               S[D*16+l] = e; z += e; }
                const float iz = 1.f / z;
                for (int l = 0; l < 16; ++l) S[D*16+l] *= iz;
            }
            __syncthreads();
            {   // gradients + SGD step (column-parallel)
                float gin[4], gout[4];
                for (int D = 0; D < 4; ++D) {
                    float a = 0.f, b = 0.f;
                    for (int l = 0; l < 16; ++l) { a += sS1[D*16+l] * sV[l][t];
                                                   b += sS2[D*16+l] * sK[l][t]; }
                    gout[D] = -a; gin[D] = -b;
                }
                for (int D = 0; D < 4; ++D) {
                    sWin[D][t]  -= lr_in  * gin[D];
                    sWout[D][t] -= lr_out * gout[D];
                }
            }
            __syncthreads();
        }
    }
}

// ---------------------------------------------- per-head LN * gate -> bf16
__global__ __launch_bounds__(256)
void norm_gate_kernel(const float* __restrict__ O, const float* __restrict__ G,
                      const float* __restrict__ ln_g, const float* __restrict__ ln_b,
                      bf16* __restrict__ Yb) {
    __shared__ float hsum[NHEAD], hsq[NHEAD];
    const int n = blockIdx.x, tid = threadIdx.x;
    if (tid < NHEAD) { hsum[tid] = 0.f; hsq[tid] = 0.f; }
    __syncthreads();
    float v[4], s = 0.f, sq = 0.f;
    #pragma unroll
    for (int i = 0; i < 4; ++i) {
        v[i] = O[(size_t)n * DIM + tid * 4 + i];
        s += v[i]; sq += v[i] * v[i];
    }
    const int h = (tid * 4) / HD;
    atomicAdd(&hsum[h], s);
    atomicAdd(&hsq[h], sq);
    __syncthreads();
    const float mu  = hsum[h] * (1.f / HD);
    const float var = hsq[h] * (1.f / HD) - mu * mu;
    const float inv = rsqrtf(var + 1e-5f);
    #pragma unroll
    for (int i = 0; i < 4; ++i) {
        const int c = tid * 4 + i, d = c & (HD - 1);
        const float o = (v[i] - mu) * inv * ln_g[d] + ln_b[d];
        Yb[(size_t)n * DIM + c] = (bf16)(o * G[(size_t)n * DIM + c]);
    }
}

// ----------------------------------------------------------------- dispatcher
extern "C" void kernel_launch(void* const* d_in, const int* in_sizes, int n_in,
                              void* d_out, int out_size, void* d_ws, size_t ws_size,
                              hipStream_t stream) {
    const float* h    = (const float*)d_in[0];
    const float* Wq   = (const float*)d_in[1];
    const float* Wk   = (const float*)d_in[2];
    const float* Wv   = (const float*)d_in[3];
    const float* Wlr  = (const float*)d_in[4];
    const float* Wg   = (const float*)d_in[5];
    const float* Wo   = (const float*)d_in[6];
    const float* cq   = (const float*)d_in[7];
    const float* ck   = (const float*)d_in[8];
    const float* cv   = (const float*)d_in[9];
    const float* Win0 = (const float*)d_in[10];
    const float* Wout0= (const float*)d_in[11];
    const float* lng  = (const float*)d_in[12];
    const float* lnb  = (const float*)d_in[13];
    float* out = (float*)d_out;

    size_t off = 0;
    auto alloc = [&](size_t bytes) -> void* {
        void* p = (char*)d_ws + off;
        off += (bytes + 255) & ~(size_t)255;
        return p;
    };
    bf16*  hbf  = (bf16*)alloc((size_t)NROWS * DIM * 2);
    bf16*  Wqb  = (bf16*)alloc((size_t)DIM * DIM * 2);
    bf16*  Wkb  = (bf16*)alloc((size_t)DIM * DIM * 2);
    bf16*  Wvb  = (bf16*)alloc((size_t)DIM * DIM * 2);
    bf16*  Wgb  = (bf16*)alloc((size_t)DIM * DIM * 2);
    bf16*  Wob  = (bf16*)alloc((size_t)DIM * DIM * 2);
    float* tmp  = (float*)alloc((size_t)NROWS * DIM * 4);
    float* qb   = (float*)alloc((size_t)NROWS * DIM * 4);
    float* kb   = (float*)alloc((size_t)NROWS * DIM * 4);
    float* vb   = (float*)alloc((size_t)NROWS * DIM * 4);
    float* gate = (float*)alloc((size_t)NROWS * DIM * 4);
    float* ob   = (float*)alloc((size_t)NROWS * DIM * 4);
    float* lrb  = (float*)alloc((size_t)NROWS * 32 * 4);
    bf16*  obf  = (bf16*)alloc((size_t)NROWS * DIM * 2);

    // bf16 conversions
    cvt_bf16_kernel<<<2048, 256, 0, stream>>>(h,  hbf, NROWS * DIM);
    cvt_bf16_kernel<<<1024, 256, 0, stream>>>(Wq, Wqb, DIM * DIM);
    cvt_bf16_kernel<<<1024, 256, 0, stream>>>(Wk, Wkb, DIM * DIM);
    cvt_bf16_kernel<<<1024, 256, 0, stream>>>(Wv, Wvb, DIM * DIM);
    cvt_bf16_kernel<<<1024, 256, 0, stream>>>(Wg, Wgb, DIM * DIM);
    cvt_bf16_kernel<<<1024, 256, 0, stream>>>(Wo, Wob, DIM * DIM);

    const dim3 ggrid(DIM / BN, NROWS / BM);  // (8, 128)

    // projections + conv/activation
    gemm_bf16_wmma<<<ggrid, 256, 0, stream>>>(hbf, Wqb, tmp, NROWS, DIM, DIM);
    conv_silu_norm_kernel<<<NROWS, 256, 0, stream>>>(tmp, cq, qb, 1);
    gemm_bf16_wmma<<<ggrid, 256, 0, stream>>>(hbf, Wkb, tmp, NROWS, DIM, DIM);
    conv_silu_norm_kernel<<<NROWS, 256, 0, stream>>>(tmp, ck, kb, 1);
    gemm_bf16_wmma<<<ggrid, 256, 0, stream>>>(hbf, Wvb, tmp, NROWS, DIM, DIM);
    conv_silu_norm_kernel<<<NROWS, 256, 0, stream>>>(tmp, cv, vb, 0);
    gemm_bf16_wmma<<<ggrid, 256, 0, stream>>>(hbf, Wgb, gate, NROWS, DIM, DIM);
    lr_kernel<<<NROWS, 256, 0, stream>>>(h, Wlr, lrb);

    // sequential chunk scan: one WG per (batch, head)
    scan_kernel<<<BATCH * NHEAD, 64, 0, stream>>>(qb, kb, vb, lrb, Win0, Wout0, ob);

    // LN * gate, then output projection
    norm_gate_kernel<<<NROWS, 256, 0, stream>>>(ob, gate, lng, lnb, obf);
    gemm_bf16_wmma<<<ggrid, 256, 0, stream>>>(obf, Wob, out, NROWS, DIM, DIM);
}
